// ReinforcemerRepacker_30073361006811
// MI455X (gfx1250) — compile-verified, matrix operand order checked by reading, not run on hardware
//
#include <hip/hip_runtime.h>
#include <hip/hip_bf16.h>
#include <math.h>

#define NN 6000
#define EE 120000
#define DD 128
#define CC 384
#define HH 4

typedef __attribute__((ext_vector_type(16))) __bf16 v16bf;
typedef __attribute__((ext_vector_type(8)))  __bf16 v8bf;
typedef __attribute__((ext_vector_type(8)))  float  v8f;
typedef __attribute__((ext_vector_type(4)))  int    v4i;

#if defined(__has_builtin)
# if __has_builtin(__builtin_amdgcn_global_load_async_to_lds_b128)
#  define ASYNC_LDS 1
# endif
# if __has_builtin(__builtin_amdgcn_s_wait_asynccnt)
#  define HAVE_WAIT_ASYNC 1
# endif
#endif
#ifndef ASYNC_LDS
# define ASYNC_LDS 0
#endif
#ifndef HAVE_WAIT_ASYNC
# define HAVE_WAIT_ASYNC 0
#endif

// ---------------------------------------------------------------------------
// Fragment load (global or LDS base): row-major bf16, 16x32 A-style fragment.
// Per CDNA5 ISA 16-bit A layout: lanes 0-15 -> K {0..7,16..23},
// lanes 16-31 -> K {8..15,24..31}; each lane holds 16 bf16 of its row.
// ---------------------------------------------------------------------------
__device__ __forceinline__ v16bf load_frag(const __bf16* base, int ld, int lane, int k0) {
  int r  = lane & 15;
  int kb = k0 + ((lane & 16) >> 1);            // +0 (lanes 0-15) or +8 (lanes 16-31)
  const __bf16* p = base + (size_t)r * ld + kb;
  v8bf lo = *reinterpret_cast<const v8bf*>(p);
  v8bf hi = *reinterpret_cast<const v8bf*>(p + 16);
  v16bf f;
#pragma unroll
  for (int i = 0; i < 8; ++i) { f[i] = lo[i]; f[i + 8] = hi[i]; }
  return f;
}

// Stage one 16-byte chunk of a weight panel row into LDS (async when available).
__device__ __forceinline__ void stage_chunk(const __bf16* g, __bf16* l) {
#if ASYNC_LDS
  __builtin_amdgcn_global_load_async_to_lds_b128(
      (__attribute__((address_space(1))) v4i*)g,
      (__attribute__((address_space(3))) v4i*)l, 0, 0);
#else
  *reinterpret_cast<v8bf*>(l) = *reinterpret_cast<const v8bf*>(g);
#endif
}

__device__ __forceinline__ void stage_fence() {
#if ASYNC_LDS
# if HAVE_WAIT_ASYNC
  __builtin_amdgcn_s_wait_asynccnt(0);
# else
  asm volatile("s_wait_asynccnt 0x0" ::: "memory");
# endif
#endif
  __syncthreads();
}

// Cooperative stage of a [128 x K] bf16 panel (rows of Wt starting at col0)
// into LDS with padded row stride ldl = K+8 (bank-conflict-free fragments).
__device__ __forceinline__ void stage_panel(const __bf16* Wt, __bf16* smem,
                                            int col0, int K, int ldl) {
  int tc = threadIdx.x & 31;        // chunk index within row, stride 32
  int tg = threadIdx.x >> 5;        // row group, stride 8
  int K8 = K >> 3;                  // 16-byte chunks per row
  for (int c = tg; c < 128; c += 8)
    for (int ch = tc; ch < K8; ch += 32)
      stage_chunk(Wt + (size_t)(col0 + c) * K + (size_t)ch * 8,
                  smem + (size_t)c * ldl + (size_t)ch * 8);
}

__device__ __forceinline__ float gelu_f(float x) {
  return 0.5f * x * (1.0f + erff(x * 0.70710678118654752f));
}

__device__ __forceinline__ void atomicMaxF(float* addr, float val) {
  unsigned int* u = reinterpret_cast<unsigned int*>(addr);
  unsigned int old = __float_as_uint(*addr);
  while (__uint_as_float(old) < val) {
    unsigned int assumed = old;
    old = atomicCAS(u, assumed, __float_as_uint(val));
    if (old == assumed) break;
  }
}

// ---------------------------------------------------------------------------
// Generic bf16 GEMM: out[M,N] = act(A[M,K] @ W[K,N] + bias); W pre-packed
// transposed as Wt[N,K] bf16. Weight panel staged in LDS, shared by 8 waves.
// One wave -> 16 rows x 128 cols (8 WMMA tiles). blockDim 256,
// grid = (ceil(M/128), N/128), dynamic LDS = (K+8)*128*2 B. act: 0=none,1=gelu.
// ---------------------------------------------------------------------------
__global__ void __launch_bounds__(256)
gemm_bf16_kernel(const __bf16* __restrict__ A, const __bf16* __restrict__ Wt,
                 const float* __restrict__ bias,
                 float* __restrict__ outF, __bf16* __restrict__ outB,
                 int M, int N, int K, int act) {
  extern __shared__ __bf16 smem[];
  const int ldl  = K + 8;
  const int col0 = blockIdx.y * 128;

  stage_panel(Wt, smem, col0, K, ldl);
  stage_fence();

  int wave = threadIdx.x >> 5;
  int lane = threadIdx.x & 31;
  int row0 = (blockIdx.x * 8 + wave) * 16;
  if (row0 >= M) return;                        // after the barrier: safe

  v8f acc[8] = {};
  const __bf16* Abase = A + (size_t)row0 * K;
  for (int k = 0; k < K; k += 32) {
    v16bf a = load_frag(Abase, K, lane, k);
#pragma unroll
    for (int t = 0; t < 8; ++t) {
      v16bf b = load_frag(smem + (size_t)(t * 16) * ldl, ldl, lane, k);
      acc[t] = __builtin_amdgcn_wmma_f32_16x16x32_bf16(false, a, false, b,
                                                       (short)0, acc[t], false, false);
    }
  }

  int cIn   = lane & 15;
  int rbase = (lane >> 4) << 3;                 // rows r (lanes 0-15) / 8+r (16-31)
#pragma unroll
  for (int t = 0; t < 8; ++t) {
    int col  = col0 + t * 16 + cIn;
    float bv = bias ? bias[col] : 0.0f;
#pragma unroll
    for (int r = 0; r < 8; ++r) {
      int row = row0 + rbase + r;
      float v = acc[t][r] + bv;
      if (act == 1) v = gelu_f(v);
      size_t o = (size_t)row * N + col;
      if (outF) outF[o] = v;
      if (outB) outB[o] = (__bf16)v;
    }
  }
}

// ---------------------------------------------------------------------------
// Attention scores with fused epilogue:
//   scores[e,h] = sum_d lrelu((feat@aW)[e,h,d]+aWb[h,d])*aAw[h,d] + aAb[h]
// grid = (ceil(E/128), H); blockDim 256 (8 waves x 16 rows); head = blockIdx.y.
// Head's 128-col weight panel staged in LDS. dyn LDS = (C+8)*128*2 B.
// ---------------------------------------------------------------------------
__global__ void __launch_bounds__(256)
att_score_kernel(const __bf16* __restrict__ feat, const __bf16* __restrict__ aWt,
                 const float* __restrict__ aWb, const float* __restrict__ aAw,
                 const float* __restrict__ aAb, float* __restrict__ scores, int M) {
  extern __shared__ __bf16 smem[];
  const int ldl = CC + 8;
  const int h   = blockIdx.y;

  stage_panel(aWt, smem, h * 128, CC, ldl);
  stage_fence();

  int wave = threadIdx.x >> 5;
  int lane = threadIdx.x & 31;
  int row0 = (blockIdx.x * 8 + wave) * 16;
  if (row0 >= M) return;

  v8f acc[8] = {};
  const __bf16* Abase = feat + (size_t)row0 * CC;
  for (int k = 0; k < CC; k += 32) {
    v16bf a = load_frag(Abase, CC, lane, k);
#pragma unroll
    for (int t = 0; t < 8; ++t) {
      v16bf b = load_frag(smem + (size_t)(t * 16) * ldl, ldl, lane, k);
      acc[t] = __builtin_amdgcn_wmma_f32_16x16x32_bf16(false, a, false, b,
                                                       (short)0, acc[t], false, false);
    }
  }

  int cIn   = lane & 15;
  int rbase = (lane >> 4) << 3;
  float part[8];
#pragma unroll
  for (int r = 0; r < 8; ++r) part[r] = 0.0f;
#pragma unroll
  for (int t = 0; t < 8; ++t) {
    int d   = t * 16 + cIn;
    float b = aWb[h * 128 + d];
    float w = aAw[h * 128 + d];
#pragma unroll
    for (int r = 0; r < 8; ++r) {
      float v = acc[t][r] + b;
      v = (v > 0.0f) ? v : 0.2f * v;            // LeakyReLU(0.2)
      part[r] += v * w;
    }
  }
#pragma unroll
  for (int m = 8; m >= 1; m >>= 1) {            // reduce within each 16-lane half
#pragma unroll
    for (int r = 0; r < 8; ++r) part[r] += __shfl_xor(part[r], m, 32);
  }
  if (cIn == 0) {
    float ab = aAb[h];
#pragma unroll
    for (int r = 0; r < 8; ++r) {
      int row = row0 + rbase + r;
      scores[(size_t)row * HH + h] = part[r] + ab;
    }
  }
}

// ---------------------------------------------------------------------------
// Small helper kernels
// ---------------------------------------------------------------------------
__global__ void pack2d_kernel(const float* __restrict__ W, __bf16* __restrict__ Wt,
                              int K, int N) {            // W[K,N] -> Wt[N,K] bf16
  int i = blockIdx.x * blockDim.x + threadIdx.x;
  if (i >= K * N) return;
  int n = i / K, k = i - n * K;
  Wt[i] = (__bf16)W[(size_t)k * N + n];
}

__global__ void pack_aw_kernel(const float* __restrict__ aW, __bf16* __restrict__ Wt) {
  // aW[H,C,D] -> Wt[(h*128+d)*C + c]
  int i = blockIdx.x * blockDim.x + threadIdx.x;
  if (i >= 512 * CC) return;
  int col = i / CC, c = i - col * CC;
  int h = col >> 7, d = col & 127;
  Wt[i] = (__bf16)aW[((size_t)h * CC + c) * DD + d];
}

__global__ void embed_kernel(const int* __restrict__ seq, const float* __restrict__ emb,
                             float* __restrict__ nodes, __bf16* __restrict__ nodesB) {
  int n = blockIdx.x, t = threadIdx.x;
  float v = emb[(size_t)seq[n] * DD + t];
  nodes[(size_t)n * DD + t]  = v;
  nodesB[(size_t)n * DD + t] = (__bf16)v;
}

__global__ void rbf_kernel(const float* __restrict__ dist, const float* __restrict__ W,
                           const float* __restrict__ b,
                           float* __restrict__ eattr, __bf16* __restrict__ eattrB) {
  __shared__ float rbf[16];
  int e = blockIdx.x, t = threadIdx.x;            // 128 threads
  if (t < 16) {
    float mu  = 2.0f + (22.0f - 2.0f) * (float)t / 15.0f;   // linspace(2,22,16)
    float sig = (22.0f - 2.0f) / 16.0f;                     // 1.25
    float x   = (dist[e] - mu) / sig;
    rbf[t] = expf(-x * x) + 1e-8f;
  }
  __syncthreads();
  float s = b[t];
#pragma unroll
  for (int j = 0; j < 16; ++j) s += rbf[j] * W[j * DD + t];
  eattr[(size_t)e * DD + t]  = s;
  eattrB[(size_t)e * DD + t] = (__bf16)s;
}

__global__ void gather_feat_kernel(const __bf16* __restrict__ nodesB,
                                   const __bf16* __restrict__ eattrB,
                                   const int* __restrict__ src, const int* __restrict__ snk,
                                   __bf16* __restrict__ feat) {
  int e = blockIdx.x, t = threadIdx.x;            // 128 threads
  size_t base = (size_t)e * CC;
  feat[base + t]          = nodesB[(size_t)src[e] * DD + t];
  feat[base + DD + t]     = nodesB[(size_t)snk[e] * DD + t];
  feat[base + 2 * DD + t] = eattrB[(size_t)e * DD + t];
}

__global__ void fill_f32_kernel(float* p, float v, int n) {
  int i = blockIdx.x * blockDim.x + threadIdx.x;
  if (i < n) p[i] = v;
}

__global__ void f32_to_bf16_kernel(const float* __restrict__ s, __bf16* __restrict__ d, int n) {
  int i = blockIdx.x * blockDim.x + threadIdx.x;
  if (i < n) d[i] = (__bf16)s[i];
}

__global__ void seg_max_kernel(const float* __restrict__ scores, const int* __restrict__ snk,
                               float* __restrict__ smax) {
  int i = blockIdx.x * blockDim.x + threadIdx.x;
  if (i >= EE * HH) return;
  int e = i >> 2, h = i & 3;
  atomicMaxF(&smax[(size_t)snk[e] * HH + h], scores[i]);
}

__global__ void att_exp_kernel(const float* __restrict__ scores, const float* __restrict__ smax,
                               const int* __restrict__ snk,
                               float* __restrict__ att, float* __restrict__ norm) {
  int i = blockIdx.x * blockDim.x + threadIdx.x;
  if (i >= EE * HH) return;
  int e = i >> 2, h = i & 3;
  float a = expf(scores[i] - smax[(size_t)snk[e] * HH + h]);
  att[i] = a;
  atomicAdd(&norm[(size_t)snk[e] * HH + h], a + 1e-12f);
}

__global__ void aggregate_kernel(const float* __restrict__ att, const float* __restrict__ norm,
                                 const float* __restrict__ nupd, const int* __restrict__ snk,
                                 float* __restrict__ agg) {
  int e = blockIdx.x, t = threadIdx.x;            // 512 threads: col = h*128+d
  int h = t >> 7, d = t & 127;
  int n = snk[e];
  float a = att[(size_t)e * HH + h] / norm[(size_t)n * HH + h];
  atomicAdd(&agg[(size_t)n * 512 + t], a * nupd[(size_t)e * DD + d]);
}

// out = LN(a + b) * g + beta; one wave per row of 128. blockDim 256 (8 rows).
__global__ void ln_residual_kernel(const float* __restrict__ a, const float* __restrict__ b,
                                   const float* __restrict__ g, const float* __restrict__ beta,
                                   float* __restrict__ out, __bf16* __restrict__ outB,
                                   int rows) {
  int row  = blockIdx.x * 8 + (threadIdx.x >> 5);
  if (row >= rows) return;
  int lane = threadIdx.x & 31;
  const float* pa = a + (size_t)row * DD;
  const float* pb = b + (size_t)row * DD;
  float x[4], s = 0.0f;
#pragma unroll
  for (int j = 0; j < 4; ++j) { x[j] = pa[lane + 32 * j] + pb[lane + 32 * j]; s += x[j]; }
#pragma unroll
  for (int m = 16; m >= 1; m >>= 1) s += __shfl_xor(s, m, 32);
  float mean = s * (1.0f / 128.0f);
  float v = 0.0f;
#pragma unroll
  for (int j = 0; j < 4; ++j) { float d = x[j] - mean; v += d * d; }
#pragma unroll
  for (int m = 16; m >= 1; m >>= 1) v += __shfl_xor(v, m, 32);
  float inv = rsqrtf(v * (1.0f / 128.0f) + 1e-5f);
#pragma unroll
  for (int j = 0; j < 4; ++j) {
    int c = lane + 32 * j;
    float o = (x[j] - mean) * inv * g[c] + beta[c];
    out[(size_t)row * DD + c] = o;
    if (outB) outB[(size_t)row * DD + c] = (__bf16)o;
  }
}

// ---------------------------------------------------------------------------
// Host launcher
// ---------------------------------------------------------------------------
extern "C" void kernel_launch(void* const* d_in, const int* in_sizes, int n_in,
                              void* d_out, int out_size, void* d_ws, size_t ws_size,
                              hipStream_t stream) {
  const int*   seq_idx   = (const int*)  d_in[0];
  const int*   edge_idx  = (const int*)  d_in[1];
  const float* edge_dist = (const float*)d_in[2];
  const float* seq_embed = (const float*)d_in[3];
  const float* elW       = (const float*)d_in[4];
  const float* elB       = (const float*)d_in[5];
  const float* aW_W      = (const float*)d_in[6];
  const float* aW_b      = (const float*)d_in[7];
  const float* aA_W      = (const float*)d_in[8];
  const float* aA_b      = (const float*)d_in[9];
  const float* nW1 = (const float*)d_in[10]; const float* nb1 = (const float*)d_in[11];
  const float* nW2 = (const float*)d_in[12]; const float* nb2 = (const float*)d_in[13];
  const float* nW3 = (const float*)d_in[14]; const float* nb3 = (const float*)d_in[15];
  const float* dW1 = (const float*)d_in[16]; const float* db1 = (const float*)d_in[17];
  const float* dW2 = (const float*)d_in[18]; const float* db2 = (const float*)d_in[19];
  const float* eW1 = (const float*)d_in[20]; const float* eb1 = (const float*)d_in[21];
  const float* eW2 = (const float*)d_in[22]; const float* eb2 = (const float*)d_in[23];
  const float* eW3 = (const float*)d_in[24]; const float* eb3 = (const float*)d_in[25];
  const float* agW = (const float*)d_in[26]; const float* agB = (const float*)d_in[27];
  const float* n1g = (const float*)d_in[28]; const float* n1b = (const float*)d_in[29];
  const float* eg  = (const float*)d_in[30]; const float* ebt = (const float*)d_in[31];

  const int* src = edge_idx;
  const int* snk = edge_idx + EE;

  // ---- workspace arena ----
  size_t off = 0;
  auto arena = [&](size_t bytes) -> void* {
    off = (off + 255) & ~(size_t)255;
    void* p = (char*)d_ws + off;
    off += bytes;
    return p;
  };
  float*  nodes    = (float*) arena((size_t)NN * DD * 4);
  __bf16* nodesB   = (__bf16*)arena((size_t)NN * DD * 2);
  float*  eattr    = (float*) arena((size_t)EE * DD * 4);
  __bf16* eattrB   = (__bf16*)arena((size_t)EE * DD * 2);
  __bf16* feat     = (__bf16*)arena((size_t)EE * CC * 2);
  __bf16* t1       = (__bf16*)arena((size_t)EE * DD * 2);
  __bf16* t2       = (__bf16*)arena((size_t)EE * DD * 2);
  float*  nupd     = (float*) arena((size_t)EE * DD * 4);
  float*  scores   = (float*) arena((size_t)EE * HH * 4);
  float*  att      = (float*) arena((size_t)EE * HH * 4);
  float*  smax     = (float*) arena((size_t)NN * HH * 4);
  float*  norm     = (float*) arena((size_t)NN * HH * 4);
  float*  agg      = (float*) arena((size_t)NN * 512 * 4);
  __bf16* aggBf    = (__bf16*)arena((size_t)NN * 512 * 2);
  float*  upd      = (float*) arena((size_t)NN * DD * 4);
  __bf16* denseT   = (__bf16*)arena((size_t)NN * 512 * 2);
  float*  dense    = (float*) arena((size_t)NN * DD * 4);
  __bf16* aWt      = (__bf16*)arena((size_t)512 * CC * 2);
  __bf16* nW1t     = (__bf16*)arena((size_t)DD * CC * 2);
  __bf16* nW2t     = (__bf16*)arena((size_t)DD * DD * 2);
  __bf16* nW3t     = (__bf16*)arena((size_t)DD * DD * 2);
  __bf16* dW1t     = (__bf16*)arena((size_t)512 * DD * 2);
  __bf16* dW2t     = (__bf16*)arena((size_t)DD * 512 * 2);
  __bf16* eW1t     = (__bf16*)arena((size_t)DD * CC * 2);
  __bf16* eW2t     = (__bf16*)arena((size_t)DD * DD * 2);
  __bf16* eW3t     = (__bf16*)arena((size_t)DD * DD * 2);
  __bf16* agWt     = (__bf16*)arena((size_t)DD * 512 * 2);
  (void)ws_size; (void)in_sizes; (void)n_in; (void)out_size;

  auto gemm = [&](const __bf16* A, const __bf16* Wt, const float* bias,
                  float* oF, __bf16* oB, int M, int Ncols, int K, int act) {
    dim3 g((M + 127) / 128, Ncols / 128);
    size_t shmem = (size_t)(K + 8) * 128 * sizeof(__bf16);
    gemm_bf16_kernel<<<g, 256, shmem, stream>>>(A, Wt, bias, oF, oB, M, Ncols, K, act);
  };
  auto pack = [&](const float* W, __bf16* Wt, int K, int Ncols) {
    int cnt = K * Ncols;
    pack2d_kernel<<<(cnt + 255) / 256, 256, 0, stream>>>(W, Wt, K, Ncols);
  };

  // ---- input embeddings ----
  embed_kernel<<<NN, DD, 0, stream>>>(seq_idx, seq_embed, nodes, nodesB);
  rbf_kernel<<<EE, DD, 0, stream>>>(edge_dist, elW, elB, eattr, eattrB);

  const int EH = EE * HH;
  const size_t attShmem = (size_t)(CC + 8) * 128 * sizeof(__bf16);
  for (int l = 0; l < 2; ++l) {
    // ---- pack this layer's weights (transposed bf16) ----
    pack_aw_kernel<<<(512 * CC + 255) / 256, 256, 0, stream>>>(aW_W + (size_t)l * HH * CC * DD, aWt);
    pack(nW1 + (size_t)l * CC * DD, nW1t, CC, DD);
    pack(nW2 + (size_t)l * DD * DD, nW2t, DD, DD);
    pack(nW3 + (size_t)l * DD * DD, nW3t, DD, DD);
    pack(dW1 + (size_t)l * DD * 512, dW1t, DD, 512);
    pack(dW2 + (size_t)l * 512 * DD, dW2t, 512, DD);
    pack(eW1 + (size_t)l * CC * DD, eW1t, CC, DD);
    pack(eW2 + (size_t)l * DD * DD, eW2t, DD, DD);
    pack(eW3 + (size_t)l * DD * DD, eW3t, DD, DD);
    pack(agW + (size_t)l * 512 * DD, agWt, 512, DD);

    // ---- feat = [nodes[src], nodes[snk], eattr] ----
    gather_feat_kernel<<<EE, DD, 0, stream>>>(nodesB, eattrB, src, snk, feat);

    // ---- attention scores (head = grid.y, weight panel staged in LDS) ----
    att_score_kernel<<<dim3((EE + 127) / 128, HH), 256, attShmem, stream>>>(
        feat, aWt, aW_b + (size_t)l * HH * DD, aA_W + (size_t)l * HH * DD,
        aA_b + (size_t)l * HH, scores, EE);

    // ---- node message MLP ----
    gemm(feat, nW1t, nb1 + (size_t)l * DD, nullptr, t1, EE, DD, CC, 1);
    gemm(t1,   nW2t, nb2 + (size_t)l * DD, nullptr, t2, EE, DD, DD, 1);
    gemm(t2,   nW3t, nb3 + (size_t)l * DD, nupd, nullptr, EE, DD, DD, 0);

    // ---- scatter softmax over sink nodes ----
    fill_f32_kernel<<<(NN * HH + 255) / 256, 256, 0, stream>>>(smax, -3.0e38f, NN * HH);
    fill_f32_kernel<<<(NN * HH + 255) / 256, 256, 0, stream>>>(norm, 0.0f, NN * HH);
    fill_f32_kernel<<<(NN * 512 + 255) / 256, 256, 0, stream>>>(agg, 0.0f, NN * 512);
    seg_max_kernel<<<(EH + 255) / 256, 256, 0, stream>>>(scores, snk, smax);
    att_exp_kernel<<<(EH + 255) / 256, 256, 0, stream>>>(scores, smax, snk, att, norm);
    aggregate_kernel<<<EE, 512, 0, stream>>>(att, norm, nupd, snk, agg);

    // ---- upd = agg @ aggr_W + b; nodes = LN(nodes + upd) ----
    f32_to_bf16_kernel<<<(NN * 512 + 255) / 256, 256, 0, stream>>>(agg, aggBf, NN * 512);
    gemm(aggBf, agWt, agB + (size_t)l * DD, upd, nullptr, NN, DD, 512, 0);
    ln_residual_kernel<<<(NN + 7) / 8, 256, 0, stream>>>(nodes, upd,
        n1g + (size_t)l * DD, n1b + (size_t)l * DD, nodes, nodesB, NN);

    // ---- dense FFN; nodes = LN(dense + upd)  (faithful residual reuse) ----
    gemm(nodesB, dW1t, db1 + (size_t)l * 512, nullptr, denseT, NN, 512, DD, 1);
    gemm(denseT, dW2t, db2 + (size_t)l * DD, dense, nullptr, NN, DD, 512, 0);
    ln_residual_kernel<<<(NN + 7) / 8, 256, 0, stream>>>(dense, upd,
        n1g + (size_t)l * DD, n1b + (size_t)l * DD, nodes, nodesB, NN);

    // ---- edge MLP; eattr = LN(eattr + eu) ----
    gather_feat_kernel<<<EE, DD, 0, stream>>>(nodesB, eattrB, src, snk, feat);
    gemm(feat, eW1t, eb1 + (size_t)l * DD, nullptr, t1, EE, DD, CC, 1);
    gemm(t1,   eW2t, eb2 + (size_t)l * DD, nullptr, t2, EE, DD, DD, 1);
    gemm(t2,   eW3t, eb3 + (size_t)l * DD, nupd, nullptr, EE, DD, DD, 0);
    ln_residual_kernel<<<(EE + 7) / 8, 256, 0, stream>>>(eattr, nupd,
        eg + (size_t)l * DD, ebt + (size_t)l * DD, eattr, eattrB, EE);
  }

  // ---- outputs: nodes [N,D] then eattr [E,De], fp32 ----
  (void)hipMemcpyAsync(d_out, nodes, (size_t)NN * DD * sizeof(float),
                       hipMemcpyDeviceToDevice, stream);
  (void)hipMemcpyAsync((float*)d_out + (size_t)NN * DD, eattr, (size_t)EE * DD * sizeof(float),
                       hipMemcpyDeviceToDevice, stream);
}
